// LstmDecoder_12661563588906
// MI455X (gfx1250) — matrix operand order repaired and checked by reading, not compiled, not run on hardware
//
#include <hip/hip_runtime.h>

// ---------------------------------------------------------------------------
// Autoregressive LSTM decoder for MI455X (gfx1250, wave32, WMMA f16->f32).
// B=128, T=512, D=256, UNITS=512, OUT=64.
//
// 8 persistent workgroups (one per 16-row batch tile), 1024 threads = 32
// waves (max WG size -> 8 waves/SIMD for maximum L2-latency hiding of the
// just-in-time weight stream). Each wave owns a 16-unit slice: 4 gates x
// 1 N-tile = 4 accumulators; LSTM cell state lives in accumulator-layout
// VGPRs (~100 VGPRs/wave total). Activations A=[x|tag|h] (16x832 f16)
// double-buffered in LDS, 2 barriers/step. Weights f16, N-major,
// L2-resident; weight loads use a laundered uniform zero offset (keeps
// global provenance -> global_load_b128, defeats LICM hoist-and-spill).
// ---------------------------------------------------------------------------

typedef __attribute__((ext_vector_type(16))) _Float16 v16h;
typedef __attribute__((ext_vector_type(8)))  _Float16 v8h;
typedef __attribute__((ext_vector_type(4)))  _Float16 v4h;
typedef __attribute__((ext_vector_type(8)))  float    v8f;
typedef __attribute__((ext_vector_type(4)))  float    v4f;

union hv16 { v16h v; v8h h[2]; };

#define BATCH   128
#define TLEN    512
#define DIN     256
#define UNITS   512
#define OUTD    64
#define AK      832      // cols: x[0,256) tag[256,320) h[320,832)
#define TAGOFF  256
#define HOFF    320
#define NGATE   2048     // 4*UNITS
#define NKT     (AK / 32)   // 26 K-tiles

__device__ __forceinline__ float fsigmoid(float x) {
    return 1.0f / (1.0f + __expf(-x));
}
__device__ __forceinline__ float ftanh(float x) {
    float e = __expf(-2.0f * x);
    return (1.0f - e) / (1.0f + e);
}

// ---------------------------------------------------------------------------
// Preprocess: f16 N-major weight matrices in workspace.
//   Wt [2048][832] : Wt[n][k] = (k<320 ? kernel[k][n] : rec_kernel[k-320][n])
//   dwt[  64][512] : dwt[n][k] = dense_w[k][n]
// ---------------------------------------------------------------------------
__global__ void prep_weights(const float* __restrict__ kern,
                             const float* __restrict__ rec,
                             const float* __restrict__ dw,
                             _Float16* __restrict__ Wt,
                             _Float16* __restrict__ dwt) {
    int idx = blockIdx.x * blockDim.x + threadIdx.x;
    const int totalW = NGATE * AK;
    if (idx < totalW) {
        int n = idx / AK, k = idx % AK;
        float v = (k < 320) ? kern[(size_t)k * NGATE + n]
                            : rec[(size_t)(k - 320) * NGATE + n];
        Wt[idx] = (_Float16)v;
    } else {
        int j = idx - totalW;
        if (j < OUTD * UNITS) {
            int n = j / UNITS, k = j % UNITS;
            dwt[j] = (_Float16)dw[(size_t)k * OUTD + n];
        }
    }
}

// ---------------------------------------------------------------------------
// Persistent decoder. grid = 8 blocks x 1024 threads (32 waves of 32).
// ---------------------------------------------------------------------------
__launch_bounds__(1024, 1)
__global__ void lstm_decoder(const float* __restrict__ x,
                             const float* __restrict__ bias,
                             const float* __restrict__ dense_b,
                             const _Float16* __restrict__ Wt,
                             const _Float16* __restrict__ dwt,
                             float* __restrict__ out) {
    __shared__ __align__(16) _Float16 Abuf[2][16 * AK];

    const int tid    = threadIdx.x;
    const int wave   = tid >> 5;      // 0..31
    const int lane   = tid & 31;
    const int ln     = lane & 15;     // tile column (B/C/D) or tile row (A)
    const int hi16   = lane >> 4;     // lane half
    const int uBase  = wave * 16;     // this wave's 16-unit slice
    const int wgRow0 = blockIdx.x * 16;

    const v8f zero8 = {0.f, 0.f, 0.f, 0.f, 0.f, 0.f, 0.f, 0.f};

    // --- init buffer 0: zero tag+h regions, load x_0 ---
    for (int i = tid; i < 16 * (AK - TAGOFF); i += 1024) {
        int r = i / (AK - TAGOFF), c = i % (AK - TAGOFF);
        Abuf[0][r * AK + TAGOFF + c] = (_Float16)0.0f;
    }
    {
        int row = tid >> 6, cg = (tid & 63) * 4;   // 64 threads per row
        const float* xp = x + (size_t)(wgRow0 + row) * (TLEN * DIN) + cg;
        v4f xv = *(const v4f*)(xp);
        v4h hv = { (_Float16)xv[0], (_Float16)xv[1],
                   (_Float16)xv[2], (_Float16)xv[3] };
        *(v4h*)&Abuf[0][row * AK + cg] = hv;
    }
    __syncthreads();

    // per-lane bias constants (column-indexed, same across lane halves)
    float bz[4];
    #pragma unroll
    for (int g = 0; g < 4; ++g)
        bz[g] = bias[g * UNITS + uBase + ln];
    float db = 0.0f;
    if (wave < 4) db = dense_b[wave * 16 + ln];

    v8f cstate = zero8;

    const int kA = hi16 * 8;    // A operand K sub-offset per lane half
    const int kB = hi16 * 8;    // B operand K sub-offset per lane half

    // Uniform ZERO element offsets, laundered per step: LICM cannot hoist
    // the weight loads across t, but pointer provenance (global) is kept.
    size_t wOff = 0, dOff = 0;

    for (int t = 0; t < TLEN; ++t) {
        _Float16* Ac = &Abuf[t & 1][0];          // read:  x_t, tag_{t-1}, h_{t-1}
        _Float16* An = &Abuf[(t + 1) & 1][0];    // write: h_t, tag_t, x_{t+1}

        asm volatile("" : "+s"(wOff), "+s"(dOff));   // defeat LICM across t
        const _Float16* WtL = Wt + wOff + (size_t)(uBase + ln) * AK + kB;

        // ---- z slice = A(16x832) @ Wt^T : 4 gates x 1 N-tile per wave ----
        v8f acc[4];
        #pragma unroll
        for (int g = 0; g < 4; ++g) acc[g] = zero8;

        for (int kt = 0; kt < NKT; ++kt) {
            hv16 a;
            const _Float16* ap = &Ac[ln * AK + kt * 32 + kA];
            a.h[0] = *(const v8h*)(ap);
            a.h[1] = *(const v8h*)(ap + 16);
            #pragma unroll
            for (int g = 0; g < 4; ++g) {
                const _Float16* bp =
                    WtL + (size_t)(g * UNITS) * AK + kt * 32;
                hv16 b;
                b.h[0] = *(const v8h*)(bp);
                b.h[1] = *(const v8h*)(bp + 16);
                acc[g] = __builtin_amdgcn_wmma_f32_16x16x32_f16(
                    false, a.v, false, b.v, (short)0, acc[g],
                    false, false);
            }
        }

        // ---- gates + cell update (registers, C/D layout) ----
        v8f hnew;
        #pragma unroll
        for (int r = 0; r < 8; ++r) {
            float zi = acc[0][r] + bz[0];
            float zf = acc[1][r] + bz[1];
            float zg = acc[2][r] + bz[2];
            float zo = acc[3][r] + bz[3];
            float ig = fsigmoid(zi);
            float fg = fsigmoid(zf);
            float gg = ftanh(zg);
            float og = fsigmoid(zo);
            float cn = fg * cstate[r] + ig * gg;
            cstate[r] = cn;
            hnew[r] = og * ftanh(cn);
        }

        // ---- write h_t into NEXT buffer (not read by anyone yet) ----
        #pragma unroll
        for (int r = 0; r < 8; ++r) {
            int m = r + hi16 * 8;
            An[m * AK + HOFF + uBase + ln] = (_Float16)hnew[r];
        }
        // ---- prefetch x_{t+1} into NEXT buffer ----
        if (t + 1 < TLEN) {
            int row = tid >> 6, cg = (tid & 63) * 4;
            const float* xp = x + (size_t)(wgRow0 + row) * (TLEN * DIN)
                                + (size_t)(t + 1) * DIN + cg;
            v4f xv = *(const v4f*)(xp);
            v4h hv = { (_Float16)xv[0], (_Float16)xv[1],
                       (_Float16)xv[2], (_Float16)xv[3] };
            *(v4h*)&An[row * AK + cg] = hv;
        }

        __syncthreads();   // h_t fully present in An

        // ---- dense: y_t = h_t(16x512) @ dwt^T, waves 0..3, 1 N-tile each --
        if (wave < 4) {
            v8f ya = zero8;
            for (int kt = 0; kt < UNITS / 32; ++kt) {
                hv16 a, b;
                const _Float16* ap = &An[ln * AK + HOFF + kt * 32 + kA];
                a.h[0] = *(const v8h*)(ap);
                a.h[1] = *(const v8h*)(ap + 16);
                const _Float16* bp =
                    dwt + dOff + (size_t)(wave * 16 + ln) * UNITS + kt * 32 + kB;
                b.h[0] = *(const v8h*)(bp);
                b.h[1] = *(const v8h*)(bp + 16);
                ya = __builtin_amdgcn_wmma_f32_16x16x32_f16(
                    false, a.v, false, b.v, (short)0, ya, false, false);
            }
            #pragma unroll
            for (int r = 0; r < 8; ++r) {
                float yv = ya[r] + db;
                int m = r + hi16 * 8;
                int ncol = wave * 16 + ln;
                out[(size_t)(wgRow0 + m) * (TLEN * OUTD)
                    + (size_t)t * OUTD + ncol] = yv;
                An[m * AK + TAGOFF + ncol] = (_Float16)yv;   // feedback tag
            }
        }

        __syncthreads();   // tag_t visible before next step's matmul
    }
}

extern "C" void kernel_launch(void* const* d_in, const int* in_sizes, int n_in,
                              void* d_out, int out_size, void* d_ws, size_t ws_size,
                              hipStream_t stream) {
    const float* x    = (const float*)d_in[0];   // [128,512,256]
    const float* kern = (const float*)d_in[1];   // [320,2048]
    const float* rec  = (const float*)d_in[2];   // [512,2048]
    const float* bias = (const float*)d_in[3];   // [2048]
    const float* dw   = (const float*)d_in[4];   // [512,64]
    const float* dbv  = (const float*)d_in[5];   // [64]

    _Float16* Wt  = (_Float16*)d_ws;                   // 2048*832 f16
    _Float16* dwt = Wt + (size_t)NGATE * AK;           // 64*512 f16

    const int totalPrep = NGATE * AK + OUTD * UNITS;
    prep_weights<<<(totalPrep + 255) / 256, 256, 0, stream>>>(kern, rec, dw, Wt, dwt);

    lstm_decoder<<<BATCH / 16, 1024, 0, stream>>>(x, bias, dbv, Wt, dwt,
                                                  (float*)d_out);
}
